// Attention_7241314861620
// MI455X (gfx1250) — matrix operand order repaired and checked by reading, not compiled, not run on hardware
//
#include <hip/hip_runtime.h>

typedef __attribute__((ext_vector_type(16))) __bf16 v16bf;
typedef __attribute__((ext_vector_type(8)))  __bf16 v8bf;
typedef __attribute__((ext_vector_type(8)))  float  v8f;

union V16 { v16bf v; v8bf h[2]; };

constexpr int Bc  = 4;
constexpr int Qc  = 512;
constexpr int Pc  = 2048;
constexpr int BSc = 16;
constexpr int Hc  = 32;
constexpr int HKc = 8;
constexpr int Dc  = 128;
constexpr int Lc  = Pc + Qc;          // 2560
constexpr int Gc  = Hc / HKc;         // 4
// SCALE * log2(e): softmax runs in the exp2 domain (v_exp_f32 is natively 2^x)
constexpr float SCALE2c = 0.08838834764831845f * 1.4426950408889634f;

constexpr int QB = 64;                // query positions per workgroup
constexpr int KC = 32;                // keys per chunk
constexpr int SK_PITCH = 136;         // bf16/row: 128 + 8 pad -> 272B = 68 banks, conflict-free b128
constexpr int VT_PITCH = 40;          // bf16/row: 32 + 8 pad  ->  80B = 20 banks

__global__ __launch_bounds__(256) void attn_prefill_kernel(
    const float* __restrict__ q,
    const float* __restrict__ k,
    const float* __restrict__ v,
    const float* __restrict__ k_cache,
    const float* __restrict__ v_cache,
    const int*   __restrict__ block_tables,
    float* __restrict__ out)
{
  // double-buffered chunk staging (~37 KB of the 320 KB LDS)
  __shared__ __align__(16) __bf16 SK [2][KC * SK_PITCH];   // K chunk [key][d]
  __shared__ __align__(16) __bf16 SVT[2][Dc * VT_PITCH];   // V chunk transposed [d][key]

  const int nqb    = Qc / QB;                 // 8
  const int wg     = blockIdx.x;
  const int qblock = wg % nqb;
  const int hk     = (wg / nqb) % HKc;
  const int b      = wg / (nqb * HKc);

  const int tid   = threadIdx.x;
  const int lane  = tid & 31;
  const int wv    = tid >> 5;                 // 0..7
  const int g     = wv >> 1;                  // 0..3 (GQA head within group)
  const int qsub  = wv & 1;                   // 0..1 (32-row sub-tile of the 64-q block)
  const int h     = hk * Gc + g;
  const int lrow  = lane & 15;
  const int lhi   = lane >> 4;                // 0/1

  const int q0 = qblock * QB + qsub * 32;     // wave's first query index (owns 32 rows: two tiles)
  int qposT[2];                               // lane's absolute position per 16-row tile
  qposT[0] = Pc + q0 + lrow;
  qposT[1] = Pc + q0 + 16 + lrow;

  // ---- Load Q rows (scaled into exp2 domain) into B-operand registers, 2 tiles x 4 d-chunks ----
  // B layout (32x16, 16-bit): lane n<16 holds K=0..7,16..23 of column n; lane n+16 holds K=8..15,24..31.
  v16bf qreg[2][4];
  #pragma unroll
  for (int T = 0; T < 2; ++T) {
    const float* qp = q + ((size_t)(b * Qc + q0 + 16 * T + lrow) * Hc + h) * Dc;
    #pragma unroll
    for (int dc = 0; dc < 4; ++dc) {
      const int dbase = dc * 32 + lhi * 8;
      const float4* q4 = (const float4*)(qp + dbase);
      #pragma unroll
      for (int j = 0; j < 4; ++j) {
        float4 f = q4[j < 2 ? j : j + 2];
        qreg[T][dc][4 * j + 0] = (__bf16)(f.x * SCALE2c);
        qreg[T][dc][4 * j + 1] = (__bf16)(f.y * SCALE2c);
        qreg[T][dc][4 * j + 2] = (__bf16)(f.z * SCALE2c);
        qreg[T][dc][4 * j + 3] = (__bf16)(f.w * SCALE2c);
      }
    }
  }

  // O^T accumulators: per tile, 8 x (16d x 16q) fp32 C-layout
  v8f accO[2][8];
  #pragma unroll
  for (int T = 0; T < 2; ++T) {
    #pragma unroll
    for (int t = 0; t < 8; ++t) {
      #pragma unroll
      for (int i = 0; i < 8; ++i) accO[T][t][i] = 0.0f;
    }
  }
  float mrow[2] = { -1e30f, -1e30f };
  float srow[2] = { 0.0f, 0.0f };

  // staging mapping: 256 threads -> 32 keys x 8 d-segments of 16 floats
  const int skey = tid >> 3;                  // 0..31
  const int dseg = (tid & 7) << 4;            // 0,16,...,112

  const int NC = Pc / KC + 2 * qblock + 2;    // chunks covering causal reach of the 64-q block

  // register prefetch buffers (software pipeline)
  float4 pkr[4], pvr[4];

  auto issue_loads = [&](int c) {
    const int pos = c * KC + skey;            // absolute key position
    const float *srcK, *srcV;
    if (pos >= Pc) {                          // new token: from k/v inputs (never mutate caches)
      const size_t r = ((size_t)(b * Qc + (pos - Pc)) * HKc + hk) * Dc;
      srcK = k + r;
      srcV = v + r;
    } else {                                  // past token: paged cache lookup
      const int   page = block_tables[b * (Lc / BSc) + (pos >> 4)];
      const size_t slot = (size_t)page * BSc + (pos & (BSc - 1));
      srcK = k_cache + (slot * HKc + hk) * Dc;
      srcV = v_cache + (slot * HKc + hk) * Dc;
    }
    const float4* k4 = (const float4*)(srcK + dseg);
    const float4* v4 = (const float4*)(srcV + dseg);
    #pragma unroll
    for (int j = 0; j < 4; ++j) { pkr[j] = k4[j]; pvr[j] = v4[j]; }
  };

  auto store_lds = [&](int buf) {
    const float* kfe = (const float*)pkr;
    const float* vfe = (const float*)pvr;
    v8bf klo, khi;
    #pragma unroll
    for (int i = 0; i < 8; ++i) {
      klo[i] = (__bf16)kfe[i];
      khi[i] = (__bf16)kfe[8 + i];
    }
    *(v8bf*)&SK[buf][skey * SK_PITCH + dseg]     = klo;
    *(v8bf*)&SK[buf][skey * SK_PITCH + dseg + 8] = khi;
    #pragma unroll
    for (int i = 0; i < 16; ++i)
      SVT[buf][(dseg + i) * VT_PITCH + skey] = (__bf16)vfe[i];
  };

  // prologue: stage chunk 0
  issue_loads(0);
  store_lds(0);
  __syncthreads();

  for (int c = 0; c < NC; ++c) {
    const int cur = c & 1;
    const bool more = (c + 1 < NC);
    if (more) issue_loads(c + 1);             // global loads overlap this chunk's compute

    // ---- S^T = K_chunk x Q^T : per tile two 16x16 sub-tiles (keys 0..15 / 16..31) ----
    // Each K A-operand load feeds BOTH q tiles -> half the LDS bytes per WMMA.
    v8f s0[2], s1[2];
    #pragma unroll
    for (int T = 0; T < 2; ++T) {
      #pragma unroll
      for (int i = 0; i < 8; ++i) { s0[T][i] = 0.0f; s1[T][i] = 0.0f; }
    }
    #pragma unroll
    for (int dc = 0; dc < 4; ++dc) {
      // A layout (16x32, 16-bit): lane<16 holds K=0..7,16..23 of row m=lane; lane>=16 K=8..15,24..31
      const int d0 = dc * 32 + lhi * 8;
      V16 a0, a1;
      a0.h[0] = *(const v8bf*)&SK[cur][lrow        * SK_PITCH + d0];
      a0.h[1] = *(const v8bf*)&SK[cur][lrow        * SK_PITCH + d0 + 16];
      a1.h[0] = *(const v8bf*)&SK[cur][(16 + lrow) * SK_PITCH + d0];
      a1.h[1] = *(const v8bf*)&SK[cur][(16 + lrow) * SK_PITCH + d0 + 16];
      #pragma unroll
      for (int T = 0; T < 2; ++T) {
        s0[T] = __builtin_amdgcn_wmma_f32_16x16x32_bf16(false, a0.v, false, qreg[T][dc], (short)0, s0[T], false, false);
        s1[T] = __builtin_amdgcn_wmma_f32_16x16x32_bf16(false, a1.v, false, qreg[T][dc], (short)0, s1[T], false, false);
      }
    }

    // ---- causal mask (only tail chunks; wave-uniform branch keeps EXEC full at WMMAs) ----
    const int kbase = c * KC;
    if (kbase + KC - 1 > Pc + q0) {
      #pragma unroll
      for (int T = 0; T < 2; ++T) {
        #pragma unroll
        for (int i = 0; i < 8; ++i) {
          const int kp0 = kbase + lhi * 8 + i;        // key of s0[T][i]
          const int kp1 = kp0 + 16;                   // key of s1[T][i]
          if (kp0 > qposT[T]) s0[T][i] = -1e30f;
          if (kp1 > qposT[T]) s1[T][i] = -1e30f;
        }
      }
    }

    // ---- online softmax in exp2 domain + P^T repack (per tile) ----
    V16 pt[2];
    #pragma unroll
    for (int T = 0; T < 2; ++T) {
      float cm = -1e30f;
      #pragma unroll
      for (int i = 0; i < 8; ++i) cm = fmaxf(cm, fmaxf(s0[T][i], s1[T][i]));
      cm = fmaxf(cm, __shfl_xor(cm, 16, 32));
      const float mnew    = fmaxf(mrow[T], cm);
      const float rescale = exp2f(mrow[T] - mnew);
      float cs = 0.0f;
      #pragma unroll
      for (int i = 0; i < 8; ++i) {
        s0[T][i] = exp2f(s0[T][i] - mnew);
        s1[T][i] = exp2f(s1[T][i] - mnew);
        cs += s0[T][i] + s1[T][i];
      }
      cs += __shfl_xor(cs, 16, 32);
      srow[T] = srow[T] * rescale + cs;
      mrow[T] = mnew;
      #pragma unroll
      for (int t = 0; t < 8; ++t) {
        #pragma unroll
        for (int i = 0; i < 8; ++i) accO[T][t][i] *= rescale;
      }
      // repack P^T (C-layout) into B-operand layout via one xor-16 fragment swap
      #pragma unroll
      for (int i = 0; i < 8; ++i) {
        const float own   = lhi ? s1[T][i] : s0[T][i];
        const float other = __shfl_xor(lhi ? s0[T][i] : s1[T][i], 16, 32);
        pt[T].v[i]     = (__bf16)(lhi ? other : own);   // B K=i   (lane<16) / K=16+i (lane>=16)
        pt[T].v[8 + i] = (__bf16)(lhi ? own : other);   // B K=8+i (lane<16) / K=24+i (lane>=16)
      }
    }

    // ---- O^T += V^T_chunk x P^T : 8 tiles across D; each V^T load feeds both q tiles ----
    #pragma unroll
    for (int t = 0; t < 8; ++t) {
      const int drow = t * 16 + lrow;
      V16 a;
      a.h[0] = *(const v8bf*)&SVT[cur][drow * VT_PITCH + lhi * 16];
      a.h[1] = *(const v8bf*)&SVT[cur][drow * VT_PITCH + lhi * 16 + 8];
      #pragma unroll
      for (int T = 0; T < 2; ++T)
        accO[T][t] = __builtin_amdgcn_wmma_f32_16x16x32_bf16(false, a.v, false, pt[T].v, (short)0, accO[T][t], false, false);
    }

    // stage chunk c+1 into the other buffer; one barrier per chunk suffices:
    // writers of buffer X are always separated from readers of X by an earlier barrier.
    if (more) store_lds((c + 1) & 1);
    __syncthreads();
  }

  // ---- normalize and write O: per tile, lane holds q-row, d = 16t + 8*lhi + i ----
  #pragma unroll
  for (int T = 0; T < 2; ++T) {
    const float inv = 1.0f / srow[T];
    float* op = out + ((size_t)(b * Qc + q0 + 16 * T + lrow) * Hc + h) * Dc;
    #pragma unroll
    for (int t = 0; t < 8; ++t) {
      const int d = t * 16 + lhi * 8;
      float4 o0, o1;
      o0.x = accO[T][t][0] * inv; o0.y = accO[T][t][1] * inv;
      o0.z = accO[T][t][2] * inv; o0.w = accO[T][t][3] * inv;
      o1.x = accO[T][t][4] * inv; o1.y = accO[T][t][5] * inv;
      o1.z = accO[T][t][6] * inv; o1.w = accO[T][t][7] * inv;
      *(float4*)(op + d)     = o0;
      *(float4*)(op + d + 4) = o1;
    }
  }
}

extern "C" void kernel_launch(void* const* d_in, const int* in_sizes, int n_in,
                              void* d_out, int out_size, void* d_ws, size_t ws_size,
                              hipStream_t stream) {
  (void)in_sizes; (void)n_in; (void)out_size; (void)d_ws; (void)ws_size;
  const float* q        = (const float*)d_in[0];
  const float* k        = (const float*)d_in[1];
  const float* v        = (const float*)d_in[2];
  const float* k_cache  = (const float*)d_in[3];
  const float* v_cache  = (const float*)d_in[4];
  const int*   bt       = (const int*)d_in[5];

  dim3 grid(Bc * HKc * (Qc / QB));   // 4 * 8 * 8 = 256 workgroups
  dim3 block(256);                   // 8 waves: (g in 0..3) x (32-row q sub-tile in 0..1)
  attn_prefill_kernel<<<grid, block, 0, stream>>>(q, k, v, k_cache, v_cache, bt, (float*)d_out);
}
